// TopKRouter_51883204935734
// MI455X (gfx1250) — matrix operand order repaired and checked by reading, not compiled, not run on hardware
//
#include <hip/hip_runtime.h>

// Problem dims (from reference)
#define B_DIM     4
#define S_DIM     4096
#define D_DIM     2048
#define E_DIM     64
#define K_TOP     2
#define M_TOTAL   (B_DIM * S_DIM)          // 16384 rows
#define KB_COUNT  (D_DIM / 32)             // 64 k-chunks of 32
#define ROWS_PER_BLOCK 128                 // 8 waves * 16 rows
#define LDS_PITCH (E_DIM + 1)              // 65 floats: bank-conflict-free column scans
#define CHUNK_HALFS (4 * 32 * 16)          // 2048 bf16 per k-chunk of swizzled W (4KB)
#define CHUNK_BYTES (CHUNK_HALFS * 2)      // 4096

typedef __attribute__((ext_vector_type(16))) __bf16       v16bf;
typedef __attribute__((ext_vector_type(8)))  float        v8f;
typedef __attribute__((ext_vector_type(4)))  unsigned int u32x4;
typedef __attribute__((ext_vector_type(8)))  int          i32x8;
typedef __attribute__((ext_vector_type(4)))  int          i32x4;
typedef __attribute__((address_space(3)))    unsigned short lds_us_t;

static __device__ __forceinline__ unsigned short f32_bf16_rne(float f) {
    unsigned int u = __float_as_uint(f);
    u += 0x7FFFu + ((u >> 16) & 1u);        // round-to-nearest-even
    return (unsigned short)(u >> 16);
}

// ---------------------------------------------------------------------------
// Kernel 1: swizzle W (f32 row-major DxE) -> bf16 in WMMA B-operand layout.
// Layout index t = (((kb*4 + nb)*32 + lane)*16 + h):
//   lane 0..15  hold N = lane,      K = kb*32 + h        (h = 0..15)
//   lane 16..31 hold N = lane-16,   K = kb*32 + 16 + h
// Each k-chunk kb occupies a contiguous 4KB block -> TDM-friendly.
// ---------------------------------------------------------------------------
__global__ void TopKRouter_swizzleW_kernel(const float* __restrict__ W,
                                           unsigned short* __restrict__ wz) {
    int t = blockIdx.x * blockDim.x + threadIdx.x;   // 0 .. D*E-1 (131072)
    if (t >= D_DIM * E_DIM) return;
    int h  = t & 15;
    int l  = (t >> 4) & 31;
    int nb = (t >> 9) & 3;
    int kb = t >> 11;
    int n  = (l & 15) + nb * 16;
    int k  = kb * 32 + ((l < 16) ? h : (16 + h));
    wz[t] = f32_bf16_rne(W[k * E_DIM + n]);
}

// ---------------------------------------------------------------------------
// TDM: DMA one contiguous 4KB k-chunk of swizzled W from global into LDS.
// D# built per CDNA5 ISA ch.8: group0 {count=1, lds_addr, global_addr, type=2},
// group1 {data_size=8B, tensor_dim0=tile_dim0=512 units, dim1=1}.
// 6-arg builtin form (amdgpu-toolchain / clang-23): extra zero group before cpol.
// ---------------------------------------------------------------------------
static __device__ __forceinline__ void tdm_load_chunk(unsigned long long gaddr,
                                                      unsigned lds_addr) {
    u32x4 g0;
    g0[0] = 1u;                                             // count=1 (valid user D#)
    g0[1] = lds_addr;                                       // lds_addr (bytes)
    g0[2] = (unsigned)(gaddr & 0xFFFFFFFFull);              // global_addr[31:0]
    g0[3] = (unsigned)((gaddr >> 32) & 0x1FFFFFFull)        // global_addr[56:32]
          | (2u << 30);                                     // type=2 ("image")

    i32x8 g1;
    g1[0] = (int)(3u << 16);          // workgroup_mask=0, data_size=3 (8 bytes)
    g1[1] = (int)(512u << 16);        // tensor_dim0[15:0]=512 (8B units = 4KB)
    g1[2] = (int)(1u << 16);          // tensor_dim0[31:16]=0, tensor_dim1[15:0]=1
    g1[3] = (int)(512u << 16);        // tensor_dim1[31:16]=0, tile_dim0=512
    g1[4] = 1;                        // tile_dim1=1, tile_dim2=0
    g1[5] = 512;                      // tensor_dim0_stride[31:0]=512
    g1[6] = (int)(512u << 16);        // stride0[47:32]=0, tensor_dim1_stride[15:0]=512
    g1[7] = 0;                        // tensor_dim1_stride[47:16]=0

    i32x4 g2 = {0, 0, 0, 0};          // tensor_dim2..: unused (<=2D)
    i32x4 g3 = {0, 0, 0, 0};
    i32x8 g4 = {0, 0, 0, 0, 0, 0, 0, 0};   // extra group (6-arg form), unused

    __builtin_amdgcn_tensor_load_to_lds(g0, g1, g2, g3, g4, 0);
}

// ---------------------------------------------------------------------------
// Kernel 2: GEMM (bf16 WMMA, f32 accum) + bias + noise + softmax + top-2.
// B staged in LDS via TDM double-buffer; A streamed from global.
// ---------------------------------------------------------------------------
struct SharedMem {
    __align__(32) unsigned short bbuf[2][CHUNK_HALFS];     // 2 x 4KB W chunks
    float logits[ROWS_PER_BLOCK][LDS_PITCH];               // epilogue staging
};

__launch_bounds__(256)
__global__ void TopKRouter_main_kernel(const float* __restrict__ x,
                                       const unsigned short* __restrict__ wz,
                                       const float* __restrict__ bias,
                                       const float* __restrict__ noise,
                                       float* __restrict__ out) {
    __shared__ SharedMem sm;

    const int tid  = threadIdx.x;
    const int wave = tid >> 5;            // 0..7
    const int lane = tid & 31;
    const int l16  = lane & 15;
    const int hsel = lane >> 4;           // 0: lanes 0-15, 1: lanes 16-31
    const int blockRow0 = blockIdx.x * ROWS_PER_BLOCK;
    const int rowBase   = blockRow0 + wave * 16;

    // LDS byte addresses of the two B staging buffers (for the TDM descriptor).
    const unsigned bb_lds0 = (unsigned)(unsigned long long)(lds_us_t*)&sm.bbuf[0][0];
    const unsigned bb_lds1 = (unsigned)(unsigned long long)(lds_us_t*)&sm.bbuf[1][0];
    const unsigned long long wz_bytes = (unsigned long long)wz;

    v8f acc0 = {}, acc1 = {}, acc2 = {}, acc3 = {};

    // A-operand addressing (16-bit A 16x32, contiguous k order matching the swizzled W):
    // lanes 0-15: row M=l16, K = {0..15}; lanes 16-31: K = {16..31}
    const float* aptr = x + (size_t)(rowBase + l16) * D_DIM + hsel * 16;

    // Prologue: TDM chunk 0 into buffer 0.
    if (wave == 0) {
        tdm_load_chunk(wz_bytes, bb_lds0);
    }
    __builtin_amdgcn_s_wait_tensorcnt(0);
    __syncthreads();

    for (int kb = 0; kb < KB_COUNT; ++kb) {
        const int cur = kb & 1;

        // Kick off DMA of the next W chunk into the idle buffer.
        if (wave == 0 && (kb + 1) < KB_COUNT) {
            tdm_load_chunk(wz_bytes + (unsigned long long)(kb + 1) * CHUNK_BYTES,
                           cur ? bb_lds0 : bb_lds1);
        }

        // A: 16 f32 -> bf16 (native f32->bf16 conversion via cast).
        const float4 f0 = *(const float4*)(aptr + 0);
        const float4 f1 = *(const float4*)(aptr + 4);
        const float4 f2 = *(const float4*)(aptr + 8);
        const float4 f3 = *(const float4*)(aptr + 12);

        v16bf a;
        a[0]  = (__bf16)f0.x;  a[1]  = (__bf16)f0.y;
        a[2]  = (__bf16)f0.z;  a[3]  = (__bf16)f0.w;
        a[4]  = (__bf16)f1.x;  a[5]  = (__bf16)f1.y;
        a[6]  = (__bf16)f1.z;  a[7]  = (__bf16)f1.w;
        a[8]  = (__bf16)f2.x;  a[9]  = (__bf16)f2.y;
        a[10] = (__bf16)f2.z;  a[11] = (__bf16)f2.w;
        a[12] = (__bf16)f3.x;  a[13] = (__bf16)f3.y;
        a[14] = (__bf16)f3.z;  a[15] = (__bf16)f3.w;

        // B: 4 N-tiles from the live LDS buffer (32B per lane per tile).
        const v16bf* bl = (const v16bf*)&sm.bbuf[cur][0];
        const v16bf b0 = bl[0 * 32 + lane];
        const v16bf b1 = bl[1 * 32 + lane];
        const v16bf b2 = bl[2 * 32 + lane];
        const v16bf b3 = bl[3 * 32 + lane];

        acc0 = __builtin_amdgcn_wmma_f32_16x16x32_bf16(false, a, false, b0,
                                                       (short)0, acc0, false, false);
        acc1 = __builtin_amdgcn_wmma_f32_16x16x32_bf16(false, a, false, b1,
                                                       (short)0, acc1, false, false);
        acc2 = __builtin_amdgcn_wmma_f32_16x16x32_bf16(false, a, false, b2,
                                                       (short)0, acc2, false, false);
        acc3 = __builtin_amdgcn_wmma_f32_16x16x32_bf16(false, a, false, b3,
                                                       (short)0, acc3, false, false);

        aptr += 32;
        if (kb + 1 < KB_COUNT) {
            __builtin_prefetch(aptr, 0, 3);   // prefetch next A chunk
        }

        // Wave 0 waits for its DMA; barrier publishes the new buffer and
        // guarantees everyone is done reading the buffer overwritten next iter.
        __builtin_amdgcn_s_wait_tensorcnt(0);
        __syncthreads();
    }

    // Scatter C tiles to LDS. C layout: VGPR j holds M = j + 8*hsel, N = l16 (+16*nb).
    const int rlo = wave * 16 + hsel * 8;
    #pragma unroll
    for (int j = 0; j < 8; ++j) {
        sm.logits[rlo + j][ 0 + l16] = acc0[j];
        sm.logits[rlo + j][16 + l16] = acc1[j];
        sm.logits[rlo + j][32 + l16] = acc2[j];
        sm.logits[rlo + j][48 + l16] = acc3[j];
    }
    __syncthreads();

    // One thread per row: bias + noise, softmax over 64, top-2, write outputs.
    if (tid < ROWS_PER_BLOCK) {
        const int R = blockRow0 + tid;

        float m  = -3.0e38f;
        float t1 = -3.0e38f, t2 = -3.0e38f;
        int   i1 = 0,        i2 = 0;

        // Pass 1: finalize logits (bias + 0.1*noise), track max and top-2 logits.
        for (int e = 0; e < E_DIM; ++e) {
            float lg = sm.logits[tid][e] + bias[e]
                     + 0.1f * noise[(size_t)R * E_DIM + e];
            sm.logits[tid][e] = lg;
            m = fmaxf(m, lg);
            if (lg > t1)       { t2 = t1; i2 = i1; t1 = lg; i1 = e; }
            else if (lg > t2)  { t2 = lg; i2 = e; }
        }

        // Pass 2: exp-sum.
        float s = 0.0f;
        for (int e = 0; e < E_DIM; ++e) {
            s += __expf(sm.logits[tid][e] - m);
        }
        const float inv = 1.0f / s;

        // Pass 3: full score matrix.
        float* scores_out = out + 2 * (size_t)M_TOTAL * K_TOP;   // region 3
        for (int e = 0; e < E_DIM; ++e) {
            scores_out[(size_t)R * E_DIM + e] =
                __expf(sm.logits[tid][e] - m) * inv;
        }

        // Regions 1 & 2: top-k scores and indices (indices stored as float).
        out[(size_t)R * K_TOP + 0] = __expf(t1 - m) * inv;
        out[(size_t)R * K_TOP + 1] = __expf(t2 - m) * inv;
        float* idx_out = out + (size_t)M_TOTAL * K_TOP;
        idx_out[(size_t)R * K_TOP + 0] = (float)i1;
        idx_out[(size_t)R * K_TOP + 1] = (float)i2;
    }
}

extern "C" void kernel_launch(void* const* d_in, const int* in_sizes, int n_in,
                              void* d_out, int out_size, void* d_ws, size_t ws_size,
                              hipStream_t stream) {
    const float* x     = (const float*)d_in[0];
    const float* W     = (const float*)d_in[1];
    const float* b     = (const float*)d_in[2];
    const float* noise = (const float*)d_in[3];
    float* out = (float*)d_out;
    unsigned short* wz = (unsigned short*)d_ws;   // 256 KB bf16 swizzled W

    // Kernel 1: W f32 -> bf16 swizzle (131072 elements)
    TopKRouter_swizzleW_kernel<<<(D_DIM * E_DIM + 255) / 256, 256, 0, stream>>>(W, wz);

    // Kernel 2: fused GEMM (TDM-staged B) + softmax + top-2
    TopKRouter_main_kernel<<<M_TOTAL / ROWS_PER_BLOCK, 256, 0, stream>>>(
        x, wz, b, noise, out);
}
